// Moondream3Attention_25975962206338
// MI455X (gfx1250) — matrix operand order repaired and checked by reading, not compiled, not run on hardware
//
#include <hip/hip_runtime.h>
#include <hip/hip_bf16.h>
#include <math.h>

// ---- problem constants -----------------------------------------------------
#define T_    2048
#define DIM_  2048
#define H_    32
#define HD_   64
#define ROT_  32
#define P_    730
#define THETA_ 1500000.0f
#define SCALE_ 0.125f          // 64^-0.5

typedef __bf16 bf16;
typedef __attribute__((ext_vector_type(16))) __bf16 v16bf;
typedef __attribute__((ext_vector_type(8)))  __bf16 v8bf;
typedef __attribute__((ext_vector_type(8)))  float  v8f;

// ---- WMMA fragment loaders (CDNA5 wave32 layouts, CDNA5 ISA) ----
// A 16x32 bf16: lane L holds row M=L%16. half=L/16:
//   elements 0..7  -> K = k0 + half*8 .. +7        (VGPR0-3)
//   elements 8..15 -> K = k0 + 16 + half*8 .. +7   (VGPR4-7)
__device__ __forceinline__ v16bf load_a_frag(const bf16* A, int lda, int m0, int k0) {
  const int lane = threadIdx.x & 31;
  const int half = lane >> 4;
  const bf16* p = A + (size_t)(m0 + (lane & 15)) * lda + k0 + half * 8;
  v8bf lo = *(const v8bf*)(p);
  v8bf hi = *(const v8bf*)(p + 16);
  return __builtin_shufflevector(lo, hi, 0,1,2,3,4,5,6,7,8,9,10,11,12,13,14,15);
}

// B 32x16 bf16 with B[k][n] = W[n][k] for row-major W[N,K]:
// lane L holds column n=n0+L%16; elements 0..7 -> K = k0 + (L/16)*8 + e, 8..15 -> K = k0 + 16 + (L/16)*8 + (e-8).
__device__ __forceinline__ v16bf load_b_frag(const bf16* W, int ldw, int n0, int k0) {
  const int lane = threadIdx.x & 31;
  const int half = lane >> 4;
  const bf16* p = W + (size_t)(n0 + (lane & 15)) * ldw + k0 + half * 8;
  v8bf lo = *(const v8bf*)(p);
  v8bf hi = *(const v8bf*)(p + 16);
  return __builtin_shufflevector(lo, hi, 0,1,2,3,4,5,6,7,8,9,10,11,12,13,14,15);
}

__device__ __forceinline__ v8f wmma_bf16(v16bf a, v16bf b, v8f c) {
  return __builtin_amdgcn_wmma_f32_16x16x32_bf16(false, a, false, b, (short)0, c, false, false);
}

// ---- kernel 0: f32 -> bf16 cast -------------------------------------------
__global__ __launch_bounds__(256) void cast_bf16_kernel(const float* __restrict__ in,
                                                        bf16* __restrict__ out, int n) {
  int i = blockIdx.x * 256 + threadIdx.x;
  if (i < n) out[i] = (bf16)in[i];
}

// ---- kernel 1/5: C[M,N] = A[M,K] @ W[N,K]^T + bias, bf16 in / f32 out ------
// One wave computes a 32x64 output tile (2 M-tiles x 4 N-tiles, 8 accums).
// Fragment loads for step k+1 are issued before the WMMAs of step k
// (software pipeline: waits become "loadcnt <= inflight" instead of 0).
__global__ __launch_bounds__(256) void gemm_bias_kernel(
    const bf16* __restrict__ A, const bf16* __restrict__ W,
    const float* __restrict__ bias, float* __restrict__ C,
    int M, int N, int K) {
  const int wave = (blockIdx.x * blockDim.x + threadIdx.x) >> 5;
  const int lane = threadIdx.x & 31;
  const int tilesN = N >> 6;
  const int m0 = (wave / tilesN) << 5;
  const int n0 = (wave % tilesN) << 6;
  if (m0 >= M) return;

  v8f acc[2][4];
#pragma unroll
  for (int i = 0; i < 2; ++i)
#pragma unroll
    for (int j = 0; j < 4; ++j) acc[i][j] = (v8f){};

  v16bf a0 = load_a_frag(A, K, m0,      0);
  v16bf a1 = load_a_frag(A, K, m0 + 16, 0);
  v16bf b0 = load_b_frag(W, K, n0,      0);
  v16bf b1 = load_b_frag(W, K, n0 + 16, 0);
  v16bf b2 = load_b_frag(W, K, n0 + 32, 0);
  v16bf b3 = load_b_frag(W, K, n0 + 48, 0);

  for (int k0 = 0; k0 < K; k0 += 32) {
    // issue next-step fragment loads first (redundant k=0 reload on last step)
    const int kn = (k0 + 32 < K) ? k0 + 32 : 0;
    __builtin_prefetch(A + (size_t)(m0 + (lane & 15)) * K + k0 + 512, 0, 0);
    v16bf na0 = load_a_frag(A, K, m0,      kn);
    v16bf na1 = load_a_frag(A, K, m0 + 16, kn);
    v16bf nb0 = load_b_frag(W, K, n0,      kn);
    v16bf nb1 = load_b_frag(W, K, n0 + 16, kn);
    v16bf nb2 = load_b_frag(W, K, n0 + 32, kn);
    v16bf nb3 = load_b_frag(W, K, n0 + 48, kn);

    acc[0][0] = wmma_bf16(a0, b0, acc[0][0]);
    acc[1][0] = wmma_bf16(a1, b0, acc[1][0]);
    acc[0][1] = wmma_bf16(a0, b1, acc[0][1]);
    acc[1][1] = wmma_bf16(a1, b1, acc[1][1]);
    acc[0][2] = wmma_bf16(a0, b2, acc[0][2]);
    acc[1][2] = wmma_bf16(a1, b2, acc[1][2]);
    acc[0][3] = wmma_bf16(a0, b3, acc[0][3]);
    acc[1][3] = wmma_bf16(a1, b3, acc[1][3]);

    a0 = na0; a1 = na1; b0 = nb0; b1 = nb1; b2 = nb2; b3 = nb3;
  }

  const int half = lane >> 4;
  const int col  = lane & 15;
  float bj[4];
#pragma unroll
  for (int j = 0; j < 4; ++j) bj[j] = bias[n0 + j * 16 + col];
#pragma unroll
  for (int i = 0; i < 2; ++i) {
#pragma unroll
    for (int r = 0; r < 8; ++r) {
      const size_t row = (size_t)(m0 + i * 16 + r + 8 * half) * N + n0 + col;
#pragma unroll
      for (int j = 0; j < 4; ++j) C[row + j * 16] = acc[i][j][r] + bj[j];
    }
  }
}

// ---- kernel 2: tok_q/tok_v = tanh(gelu(qkv[t]) @ tau_w^T) ------------------
__global__ __launch_bounds__(256) void tok_kernel(
    const float* __restrict__ qkv, const float* __restrict__ wq,
    const float* __restrict__ wv, float* __restrict__ tokq, float* __restrict__ tokv) {
  __shared__ float g[3 * DIM_];
  const int t = blockIdx.x;
  for (int i = threadIdx.x; i < 3 * DIM_; i += 256) {
    float x = qkv[(size_t)t * 3 * DIM_ + i];
    g[i] = 0.5f * x * (1.0f + erff(x * 0.70710678118f));   // exact GELU
  }
  __syncthreads();
  const int wave = threadIdx.x >> 5, lane = threadIdx.x & 31;
#pragma unroll 1
  for (int oo = 0; oo < 8; ++oo) {
    const int o = wave * 8 + oo;                            // 0..63
    const float* w = (o < 32) ? (wq + (size_t)o * 3 * DIM_)
                              : (wv + (size_t)(o - 32) * 3 * DIM_);
    float s = 0.0f;
    for (int i = lane; i < 3 * DIM_; i += 32) s += g[i] * w[i];
    s += __shfl_xor(s, 1);  s += __shfl_xor(s, 2);
    s += __shfl_xor(s, 4);  s += __shfl_xor(s, 8);  s += __shfl_xor(s, 16);
    if (lane == 0) {
      float v = tanhf(s);
      if (o < 32) tokq[t * H_ + o] = v; else tokv[t * H_ + (o - 32)] = v;
    }
  }
}

// ---- kernel 3: tau scaling + RoPE -> Qr[h][t][d], Kr[h][t][d], Vt[h][d][t] -
__global__ __launch_bounds__(256) void prep_kernel(
    const int* __restrict__ positions, const float* __restrict__ qkv,
    const float* __restrict__ tokq, const float* __restrict__ tokv,
    const float* __restrict__ tau_alpha,
    bf16* __restrict__ Qr, bf16* __restrict__ Kr, bf16* __restrict__ Vt) {
  const int t = blockIdx.x;
  const float pos = (float)positions[t];
  const float pl  = logf(fmaxf(pos + 1.0f, 1e-6f));
  const float* row = qkv + (size_t)t * 3 * DIM_;
  for (int idx = threadIdx.x; idx < H_ * HD_; idx += 256) {
    const int h = idx >> 6, d = idx & 63;
    const float tp = 0.5f + 1.0f / (1.0f + __expf(-pl * tau_alpha[h])); // 1+sig-0.5
    const float tq = tokq[t * H_ + h] + tp;
    const float tv = tokv[t * H_ + h] + tp;
    float q = row[h * 64 + d] * tq;
    float k = row[DIM_ + h * 64 + d];
    float v = row[2 * DIM_ + h * 64 + d] * tv;
    float qr = q, kr = k;
    if (d < ROT_) {
      const int j = d & 15;
      const float inv = __powf(THETA_, -(float)j / 16.0f);
      const float ang = pos * inv;
      const float cc = __cosf(ang), ss = __sinf(ang);
      if (d < 16) {
        float q2 = row[h * 64 + d + 16] * tq;
        float k2 = row[DIM_ + h * 64 + d + 16];
        qr = q * cc - q2 * ss;  kr = k * cc - k2 * ss;
      } else {
        float q1 = row[h * 64 + d - 16] * tq;
        float k1 = row[DIM_ + h * 64 + d - 16];
        qr = q * cc + q1 * ss;  kr = k * cc + k1 * ss;
      }
    }
    const size_t hb = (size_t)h * T_ * HD_;
    Qr[hb + (size_t)t * HD_ + d] = (bf16)qr;
    Kr[hb + (size_t)t * HD_ + d] = (bf16)kr;
    Vt[(size_t)(h * HD_ + d) * T_ + t] = (bf16)v;
  }
}

// ---- kernel 4: flash attention, one wave = (head, 16 query rows) -----------
__global__ __launch_bounds__(256) void flash_kernel(
    const bf16* __restrict__ Qr, const bf16* __restrict__ Kr,
    const bf16* __restrict__ Vt, bf16* __restrict__ O) {
  __shared__ bf16 sP[8][16 * 32];
  const int wave = threadIdx.x >> 5;
  const int lane = threadIdx.x & 31;
  const int gw = blockIdx.x * 8 + wave;
  const int h  = gw >> 7;                    // 128 query tiles per head
  const int m0 = (gw & 127) << 4;
  const bf16* Qh = Qr + (size_t)h * T_ * HD_;
  const bf16* Kh = Kr + (size_t)h * T_ * HD_;
  const bf16* Vh = Vt + (size_t)h * HD_ * T_;
  const int half = lane >> 4;
  const int col  = lane & 15;

  const v16bf aq0 = load_a_frag(Qh, HD_, m0, 0);
  const v16bf aq1 = load_a_frag(Qh, HD_, m0, 32);

  float rmax[8], rsum[8];
#pragma unroll
  for (int r = 0; r < 8; ++r) { rmax[r] = -1e30f; rsum[r] = 0.0f; }
  v8f o0 = {}, o1 = {}, o2 = {}, o3 = {};

  int smax = m0 + 15;
  if (m0 < P_ && P_ - 1 > smax) smax = P_ - 1;   // P x P prefix is unmasked
  const int nchunk = (smax + 32) >> 5;           // ceil((smax+1)/32)
  bf16* myP = sP[wave];

  for (int c = 0; c < nchunk; ++c) {
    const int key0 = c << 5;
    // ---- scores for 32 keys: two 16x16 f32 tiles (K=64 in two wmma steps)
    v8f s0 = {}, s1 = {};
    s0 = wmma_bf16(aq0, load_b_frag(Kh, HD_, key0,      0), s0);
    s0 = wmma_bf16(aq1, load_b_frag(Kh, HD_, key0,     32), s0);
    s1 = wmma_bf16(aq0, load_b_frag(Kh, HD_, key0 + 16, 0), s1);
    s1 = wmma_bf16(aq1, load_b_frag(Kh, HD_, key0 + 16, 32), s1);

    const int n0k = key0 + col, n1k = key0 + 16 + col;
#pragma unroll
    for (int r = 0; r < 8; ++r) {
      const int m = m0 + r + 8 * half;
      float x0 = s0[r] * SCALE_;
      float x1 = s1[r] * SCALE_;
      const bool mrowP = (m < P_);
      if (!((n0k <= m) || (n0k < P_ && mrowP))) x0 = -1e30f;
      if (!((n1k <= m) || (n1k < P_ && mrowP))) x1 = -1e30f;
      // chunk row-max across the 16-lane group
      float v = fmaxf(x0, x1);
      v = fmaxf(v, __shfl_xor(v, 1));
      v = fmaxf(v, __shfl_xor(v, 2));
      v = fmaxf(v, __shfl_xor(v, 4));
      v = fmaxf(v, __shfl_xor(v, 8));
      const float nm = fmaxf(rmax[r], v);
      const float alpha = __expf(rmax[r] - nm);
      rmax[r] = nm;
      const float p0 = __expf(x0 - nm);
      const float p1 = __expf(x1 - nm);
      float ps = p0 + p1;
      ps += __shfl_xor(ps, 1); ps += __shfl_xor(ps, 2);
      ps += __shfl_xor(ps, 4); ps += __shfl_xor(ps, 8);
      rsum[r] = rsum[r] * alpha + ps;
      o0[r] *= alpha; o1[r] *= alpha; o2[r] *= alpha; o3[r] *= alpha;
      // stage P tile (16 rows x 32 keys) into LDS for the A-fragment re-layout
      myP[(r + 8 * half) * 32 + col]      = (bf16)p0;
      myP[(r + 8 * half) * 32 + 16 + col] = (bf16)p1;
    }
    // wave-local LDS RAW fence before the transposed fragment read
    asm volatile("s_wait_dscnt 0" ::: "memory");
    const v16bf ap = load_a_frag(myP, 32, 0, 0);
    o0 = wmma_bf16(ap, load_b_frag(Vh, T_,  0, key0), o0);
    o1 = wmma_bf16(ap, load_b_frag(Vh, T_, 16, key0), o1);
    o2 = wmma_bf16(ap, load_b_frag(Vh, T_, 32, key0), o2);
    o3 = wmma_bf16(ap, load_b_frag(Vh, T_, 48, key0), o3);
  }

#pragma unroll
  for (int r = 0; r < 8; ++r) {
    const float inv = 1.0f / rsum[r];
    const size_t base = (size_t)(m0 + r + 8 * half) * DIM_ + h * HD_ + col;
    O[base]      = (bf16)(o0[r] * inv);
    O[base + 16] = (bf16)(o1[r] * inv);
    O[base + 32] = (bf16)(o2[r] * inv);
    O[base + 48] = (bf16)(o3[r] * inv);
  }
}

// ---- host-side orchestration ----------------------------------------------
extern "C" void kernel_launch(void* const* d_in, const int* in_sizes, int n_in,
                              void* d_out, int out_size, void* d_ws, size_t ws_size,
                              hipStream_t stream) {
  (void)in_sizes; (void)n_in; (void)out_size; (void)ws_size;
  const int*   positions = (const int*)  d_in[0];
  const float* hs        = (const float*)d_in[1];
  const float* wqkv      = (const float*)d_in[2];
  const float* bqkv      = (const float*)d_in[3];
  const float* wout      = (const float*)d_in[4];
  const float* bout      = (const float*)d_in[5];
  const float* talpha    = (const float*)d_in[6];
  const float* twq       = (const float*)d_in[7];
  const float* twv       = (const float*)d_in[8];
  float* out = (float*)d_out;

  char* ws = (char*)d_ws;
  size_t off = 0;
  auto take = [&](size_t bytes) { void* p = ws + off; off += (bytes + 255) & ~(size_t)255; return p; };

  bf16*  hsb    = (bf16*) take((size_t)T_ * DIM_ * 2);
  bf16*  wqkvb  = (bf16*) take((size_t)3 * DIM_ * DIM_ * 2);
  bf16*  woutb  = (bf16*) take((size_t)DIM_ * DIM_ * 2);
  float* qkvf   = (float*)take((size_t)T_ * 3 * DIM_ * 4);
  float* tokq   = (float*)take((size_t)T_ * H_ * 4);
  float* tokv   = (float*)take((size_t)T_ * H_ * 4);
  bf16*  Qrb    = (bf16*) take((size_t)H_ * T_ * HD_ * 2);
  bf16*  Krb    = (bf16*) take((size_t)H_ * T_ * HD_ * 2);
  bf16*  Vtb    = (bf16*) take((size_t)H_ * HD_ * T_ * 2);
  bf16*  Obf    = (bf16*) take((size_t)T_ * DIM_ * 2);

  // 0) casts
  {
    int n1 = T_ * DIM_, n2 = 3 * DIM_ * DIM_, n3 = DIM_ * DIM_;
    cast_bf16_kernel<<<(n1 + 255) / 256, 256, 0, stream>>>(hs,   hsb,   n1);
    cast_bf16_kernel<<<(n2 + 255) / 256, 256, 0, stream>>>(wqkv, wqkvb, n2);
    cast_bf16_kernel<<<(n3 + 255) / 256, 256, 0, stream>>>(wout, woutb, n3);
  }
  // 1) qkv = hs @ Wqkv^T + b   (M=2048, N=6144, K=2048), 32x64 tile per wave
  {
    int waves = (T_ / 32) * (3 * DIM_ / 64);         // 6144
    gemm_bias_kernel<<<waves / 8, 256, 0, stream>>>(hsb, wqkvb, bqkv, qkvf,
                                                    T_, 3 * DIM_, DIM_);
  }
  // 2) tau token features
  tok_kernel<<<T_, 256, 0, stream>>>(qkvf, twq, twv, tokq, tokv);
  // 3) RoPE + scaling -> per-head Q, K, V^T (bf16)
  prep_kernel<<<T_, 256, 0, stream>>>(positions, qkvf, tokq, tokv, talpha,
                                      Qrb, Krb, Vtb);
  // 4) attention
  flash_kernel<<<(H_ * (T_ / 16)) / 8, 256, 0, stream>>>(Qrb, Krb, Vtb, Obf);
  // 5) out = O @ Wout^T + b    (M=2048, N=2048, K=2048)
  {
    int waves = (T_ / 32) * (DIM_ / 64);             // 2048
    gemm_bias_kernel<<<waves / 8, 256, 0, stream>>>(Obf, woutb, bout, out,
                                                    T_, DIM_, DIM_);
  }
}
